// AttentionHead_8426725834822
// MI455X (gfx1250) — matrix operand order repaired and checked by reading, not compiled, not run on hardware
//
#include <hip/hip_runtime.h>
#include <hip/hip_bf16.h>

// ---------------- CDNA5 WMMA types & helpers ----------------
typedef __attribute__((ext_vector_type(16))) __bf16 v16bf;
typedef __attribute__((ext_vector_type(8)))  __bf16 v8bf;
typedef __attribute__((ext_vector_type(8)))  float  v8f;

union ABTile { v16bf v; v8bf h[2]; };

// A-matrix tile (16x32, row-major source, leading dim lda).
// Per ISA 7.12.2: lane L -> M = L%16, baseK = (L/16)*8;
// elements 0..7 -> K = baseK+0..7, elements 8..15 -> K = baseK+16..23.
__device__ __forceinline__ v16bf load_a_16x32(const __bf16* A, int lda, int lane, int k0) {
    int m  = lane & 15;
    int kb = (lane >> 4) << 3;       // 0 or 8
    const __bf16* p = A + (size_t)m * lda + k0 + kb;
    ABTile t;
    t.h[0] = *(const v8bf*)(p);      // K = kb + 0..7   (16B aligned)
    t.h[1] = *(const v8bf*)(p + 16); // K = kb + 16..23
    return t.v;
}

// B^T tile (K=32 x N=16) built from row-major B (N x K) -> "NT" GEMM.
// Per ISA: lane L -> N = L%16, baseK = (L/16)*8; elements 0..7 -> K = baseK+0..7, 8..15 -> K = baseK+16..23.
__device__ __forceinline__ v16bf load_bT_32x16(const __bf16* B, int ldb, int lane, int k0) {
    int n  = lane & 15;
    int kb = (lane >> 4) << 3;       // 0 or 8
    const __bf16* p = B + (size_t)n * ldb + k0 + kb;
    ABTile t;
    t.h[0] = *(const v8bf*)(p);
    t.h[1] = *(const v8bf*)(p + 16);
    return t.v;
}

__device__ __forceinline__ v8f wmma_bf16(v16bf a, v16bf b, v8f c) {
    return __builtin_amdgcn_wmma_f32_16x16x32_bf16(false, a, false, b, (short)0, c,
                                                   false, false);
}

// Async global->LDS copy (gfx1250 ASYNCcnt path). ldsoff = workgroup-relative LDS
// byte address (low 32 bits of the generic shared pointer), gaddr = global address.
__device__ __forceinline__ void async_copy_b128(unsigned ldsoff, const void* gaddr) {
    asm volatile("global_load_async_to_lds_b128 %0, %1, off"
                 :: "v"(ldsoff), "v"(gaddr) : "memory");
}
__device__ __forceinline__ void wait_async0() {
    asm volatile("s_wait_asynccnt 0x0" ::: "memory");
}

// ---------------- Problem constants ----------------
#define BATCH   16
#define SEQ     1024
#define FEAT    512
#define ROWS    (BATCH * SEQ)          // 16384
#define KT32    (FEAT / 32)            // 16 k-steps of 32
#define NT16    (FEAT / 16)            // 32 n-tiles
#define SCALE   0.044194173824159216f  // 1/sqrt(512)

// ---------------- fp32 -> bf16 convert ----------------
__global__ void cvt_bf16_kernel(const float* __restrict__ src, __bf16* __restrict__ dst, int n) {
    int i = blockIdx.x * blockDim.x + threadIdx.x;
    if (i < n) dst[i] = (__bf16)src[i];
}

__global__ void zero_kernel(float* p, int n) {
    int i = blockIdx.x * blockDim.x + threadIdx.x;
    if (i < n) p[i] = 0.0f;
}

// ---------------- Stage 1: qk = X*Wqk^T, vT = (X*Wv^T)^T (per batch) ----------------
// One block per 16-row M stripe. The A stripe (16x512 bf16 = 16KB) is staged into
// LDS once via async global->LDS; each of the 8 waves computes 4 n-tiles for BOTH
// weight matrices (8 accumulators). Per k-step all 8 B tiles are loaded into
// distinct registers FIRST, then the 8 WMMAs issue, so loads overlap the matrix pipe.
__global__ void __launch_bounds__(256) gemm_qv_kernel(const __bf16* __restrict__ X,
                                                      const __bf16* __restrict__ Wqk,
                                                      const __bf16* __restrict__ Wv,
                                                      __bf16* __restrict__ qk,
                                                      __bf16* __restrict__ vT) {
    __shared__ __attribute__((aligned(16))) __bf16 sA[16 * FEAT];   // 16KB
    int tid  = threadIdx.x;
    int wave = tid >> 5;
    int lane = tid & 31;
    int mt   = blockIdx.x;                         // 0..1023
    const __bf16* Arow = X + (size_t)mt * 16 * FEAT;

    // stage A stripe: 16384 bytes = 4 passes x (256 threads x 16B)
    unsigned ldsbase = (unsigned)(size_t)(void*)sA;
#pragma unroll
    for (int p = 0; p < 4; ++p) {
        unsigned off = p * 4096 + tid * 16;
        async_copy_b128(ldsbase + off, (const char*)Arow + off);
    }
    wait_async0();
    __syncthreads();

    int nt0 = wave * 4;                            // 4 n-tiles per wave
    v8f zero = {};
    v8f acc1[4] = {zero, zero, zero, zero};
    v8f acc2[4] = {zero, zero, zero, zero};
#pragma unroll 2
    for (int kt = 0; kt < KT32; ++kt) {
        v16bf a = load_a_16x32(sA, FEAT, lane, kt * 32);
        v16bf b1[4], b2[4];
#pragma unroll
        for (int j = 0; j < 4; ++j) {
            b1[j] = load_bT_32x16(Wqk + (size_t)(nt0 + j) * 16 * FEAT, FEAT, lane, kt * 32);
            b2[j] = load_bT_32x16(Wv  + (size_t)(nt0 + j) * 16 * FEAT, FEAT, lane, kt * 32);
        }
#pragma unroll
        for (int j = 0; j < 4; ++j) {
            acc1[j] = wmma_bf16(a, b1[j], acc1[j]);
            acc2[j] = wmma_bf16(a, b2[j], acc2[j]);
        }
    }
    int n = lane & 15, mb = (lane >> 4) * 8;
#pragma unroll
    for (int j = 0; j < 4; ++j) {
        int nt = nt0 + j;
#pragma unroll
        for (int r = 0; r < 8; ++r) {
            int rowg = mt * 16 + mb + r;                   // global row 0..16383
            qk[(size_t)rowg * FEAT + nt * 16 + n] = (__bf16)acc1[j][r];
            int b = rowg >> 10;                            // batch
            int m = rowg & (SEQ - 1);                      // row in batch
            vT[((size_t)b * FEAT + nt * 16 + n) * SEQ + m] = (__bf16)acc2[j][r];
        }
    }
}

// ---------------- Stage 2: attention (scores -> softmax -> P*V) ----------------
// One wave per (batch, 16-row slab). The 16x512 qk A-slab lives in registers
// (16 k-tiles x 8 VGPRs). LDS: 16 rows x 1024 f32 scores (64KB); softmax is done
// row-at-a-time fully in registers; bf16 P overwrites each row in place;
// 1/rowsum stashed in the unused back half of each row.
__global__ void __launch_bounds__(32) attn_kernel(const __bf16* __restrict__ qk,
                                                  const __bf16* __restrict__ vT,
                                                  __bf16* __restrict__ mixed) {
    extern __shared__ char smem[];
    float* sS = (float*)smem;                 // [16][1024] f32, row stride 1024 floats
    int b    = blockIdx.x >> 6;               // 0..15
    int slab = blockIdx.x & 63;               // 0..63
    int lane = threadIdx.x;                   // 0..31 (one wave)
    int row0 = slab * 16;
    const __bf16* qkB = qk + (size_t)b * SEQ * FEAT;
    const __bf16* vTB = vT + (size_t)b * FEAT * SEQ;
    const __bf16* Arow = qkB + (size_t)row0 * FEAT;
    int n = lane & 15, mb = (lane >> 4) * 8;

    // register-resident A slab: 16 tiles x 8 VGPRs = 128 VGPRs
    v16bf areg[KT32];
#pragma unroll
    for (int kt = 0; kt < KT32; ++kt)
        areg[kt] = load_a_16x32(Arow, FEAT, lane, kt * 32);

    // --- scores: S = qk_slab (16x512) x qk^T (512x1024) ---
    for (int ct = 0; ct < SEQ / 16; ++ct) {
        v8f acc = {};
        const __bf16* Bcol = qkB + (size_t)ct * 16 * FEAT;
#pragma unroll
        for (int kg = 0; kg < KT32 / 4; ++kg) {
            v16bf bm[4];
#pragma unroll
            for (int j = 0; j < 4; ++j)
                bm[j] = load_bT_32x16(Bcol, FEAT, lane, (kg * 4 + j) * 32);
#pragma unroll
            for (int j = 0; j < 4; ++j)
                acc = wmma_bf16(areg[kg * 4 + j], bm[j], acc);
        }
#pragma unroll
        for (int r = 0; r < 8; ++r)
            sS[(size_t)(mb + r) * SEQ + ct * 16 + n] = acc[r];
    }
    __syncthreads();

    // --- softmax per row (all 32 lanes cooperate on one row at a time) ---
    for (int r = 0; r < 16; ++r) {
        float* srow = sS + (size_t)r * SEQ;
        float vals[SEQ / 32];
        float mx = -3.402823466e38f;
#pragma unroll
        for (int c = 0; c < SEQ / 32; ++c) {
            vals[c] = srow[lane + 32 * c] * SCALE;
            mx = fmaxf(mx, vals[c]);
        }
#pragma unroll
        for (int off = 16; off > 0; off >>= 1) mx = fmaxf(mx, __shfl_xor(mx, off));
        float s = 0.0f;
        __bf16* prow = (__bf16*)srow;          // bf16 P row reuses first 2KB of the row
#pragma unroll
        for (int c = 0; c < SEQ / 32; ++c) {
            float e = __expf(vals[c] - mx);
            s += e;
            prow[lane + 32 * c] = (__bf16)e;
        }
#pragma unroll
        for (int off = 16; off > 0; off >>= 1) s += __shfl_xor(s, off);
        if (lane == 0) srow[SEQ / 2] = 1.0f / s;   // stash 1/sum after the P row
    }
    __syncthreads();

    // --- mixed = P (16x1024) x V (1024x512), via vT in NT layout ---
    const __bf16* P = (const __bf16*)smem;     // row stride = 2*SEQ bf16 elements
    for (int nt = 0; nt < NT16; ++nt) {
        v8f acc = {};
        const __bf16* Bcol = vTB + (size_t)nt * 16 * SEQ;
#pragma unroll
        for (int kg = 0; kg < (SEQ / 32) / 4; ++kg) {
            v16bf am[4], bm[4];
#pragma unroll
            for (int j = 0; j < 4; ++j) {
                am[j] = load_a_16x32(P, 2 * SEQ, lane, (kg * 4 + j) * 32);
                bm[j] = load_bT_32x16(Bcol, SEQ, lane, (kg * 4 + j) * 32);
            }
#pragma unroll
            for (int j = 0; j < 4; ++j)
                acc = wmma_bf16(am[j], bm[j], acc);
        }
#pragma unroll
        for (int r = 0; r < 8; ++r) {
            float inv = sS[(size_t)(mb + r) * SEQ + SEQ / 2];
            mixed[((size_t)b * SEQ + row0 + mb + r) * FEAT + nt * 16 + n] =
                (__bf16)(acc[r] * inv);
        }
    }
}

// ---------------- Stage 3: h = mixed*Wf^T + bf, LeakyReLU, BN stats ----------------
// Same block structure as gemm_qv: A stripe async-staged in LDS, 4 n-tiles/wave,
// B tiles gathered into distinct registers before the WMMA chain.
__global__ void __launch_bounds__(256) ffn_kernel(const __bf16* __restrict__ mixed,
                                                  const __bf16* __restrict__ Wf,
                                                  const float* __restrict__ bias,
                                                  float* __restrict__ h,
                                                  float* __restrict__ fsum,
                                                  float* __restrict__ fsumsq) {
    __shared__ __attribute__((aligned(16))) __bf16 sA[16 * FEAT];   // 16KB
    int tid  = threadIdx.x;
    int wave = tid >> 5;
    int lane = tid & 31;
    int mt   = blockIdx.x;
    const __bf16* Arow = mixed + (size_t)mt * 16 * FEAT;

    unsigned ldsbase = (unsigned)(size_t)(void*)sA;
#pragma unroll
    for (int p = 0; p < 4; ++p) {
        unsigned off = p * 4096 + tid * 16;
        async_copy_b128(ldsbase + off, (const char*)Arow + off);
    }
    wait_async0();
    __syncthreads();

    int nt0 = wave * 4;
    v8f zero = {};
    v8f acc[4] = {zero, zero, zero, zero};
#pragma unroll 2
    for (int kt = 0; kt < KT32; ++kt) {
        v16bf a = load_a_16x32(sA, FEAT, lane, kt * 32);
        v16bf bm[4];
#pragma unroll
        for (int j = 0; j < 4; ++j)
            bm[j] = load_bT_32x16(Wf + (size_t)(nt0 + j) * 16 * FEAT, FEAT, lane, kt * 32);
#pragma unroll
        for (int j = 0; j < 4; ++j)
            acc[j] = wmma_bf16(a, bm[j], acc[j]);
    }
    int n = lane & 15, mb = (lane >> 4) * 8;
#pragma unroll
    for (int j = 0; j < 4; ++j) {
        int nt = nt0 + j;
        float bv = bias[nt * 16 + n];
        float s1 = 0.0f, s2 = 0.0f;
#pragma unroll
        for (int r = 0; r < 8; ++r) {
            float val = acc[j][r] + bv;
            val = (val >= 0.0f) ? val : 0.2f * val;     // LeakyReLU(0.2)
            h[((size_t)mt * 16 + mb + r) * FEAT + nt * 16 + n] = val;
            s1 += val;
            s2 += val * val;
        }
        s1 += __shfl_xor(s1, 16);
        s2 += __shfl_xor(s2, 16);
        if (lane < 16) {
            atomicAdd(&fsum[nt * 16 + n], s1);
            atomicAdd(&fsumsq[nt * 16 + n], s2);
        }
    }
}

// ---------------- Stage 4: BatchNorm apply ----------------
__global__ void bn_kernel(const float* __restrict__ h,
                          const float* __restrict__ fsum,
                          const float* __restrict__ fsumsq,
                          const float* __restrict__ gamma,
                          const float* __restrict__ beta,
                          float* __restrict__ out, int total) {
    int i = blockIdx.x * blockDim.x + threadIdx.x;
    if (i >= total) return;
    int f = i & (FEAT - 1);
    const float invN = 1.0f / (float)ROWS;
    float mean = fsum[f] * invN;
    float var  = fsumsq[f] * invN - mean * mean;
    float x = (h[i] - mean) * rsqrtf(var + 1e-5f);
    out[i] = x * gamma[f] + beta[f];
}

// ---------------- Host launcher ----------------
extern "C" void kernel_launch(void* const* d_in, const int* in_sizes, int n_in,
                              void* d_out, int out_size, void* d_ws, size_t ws_size,
                              hipStream_t stream) {
    const float* x     = (const float*)d_in[0];
    const float* Wqk   = (const float*)d_in[1];
    const float* Wv    = (const float*)d_in[2];
    const float* Wf    = (const float*)d_in[3];
    const float* bf_   = (const float*)d_in[4];
    const float* gamma = (const float*)d_in[5];
    const float* beta  = (const float*)d_in[6];

    const size_t XB = (size_t)ROWS * FEAT;   // 8,388,608 elements
    const size_t WB = (size_t)FEAT * FEAT;   //   262,144 elements

    char* ws = (char*)d_ws;
    __bf16* xb   = (__bf16*)ws;               ws += 2 * XB;
    __bf16* wqkb = (__bf16*)ws;               ws += 2 * WB;
    __bf16* wvb  = (__bf16*)ws;               ws += 2 * WB;
    __bf16* wfb  = (__bf16*)ws;               ws += 2 * WB;
    __bf16* qkb  = (__bf16*)ws;               ws += 2 * XB;
    __bf16* vTb  = (__bf16*)ws;               ws += 2 * XB;
    __bf16* mixb = (__bf16*)ws;               ws += 2 * XB;
    float*  h    = (float*)ws;                ws += 4 * XB;
    float*  fsum = (float*)ws;                ws += 4 * FEAT;
    float*  fsq  = (float*)ws;

    // convert inputs to bf16
    cvt_bf16_kernel<<<(int)(XB / 256), 256, 0, stream>>>(x, xb, (int)XB);
    cvt_bf16_kernel<<<(int)(WB / 256), 256, 0, stream>>>(Wqk, wqkb, (int)WB);
    cvt_bf16_kernel<<<(int)(WB / 256), 256, 0, stream>>>(Wv, wvb, (int)WB);
    cvt_bf16_kernel<<<(int)(WB / 256), 256, 0, stream>>>(Wf, wfb, (int)WB);
    zero_kernel<<<4, 256, 0, stream>>>(fsum, 2 * FEAT);   // fsum & fsq contiguous

    // projections (qk and v-transposed): one block per 16-row stripe
    gemm_qv_kernel<<<1024, 256, 0, stream>>>(xb, wqkb, wvb, qkb, vTb);

    // attention: 16 batches x 64 slabs, one wave each, 64KB dynamic LDS
    attn_kernel<<<BATCH * 64, 32, 16 * SEQ * (int)sizeof(float), stream>>>(qkb, vTb, mixb);

    // ffn + leaky + BN statistics
    ffn_kernel<<<1024, 256, 0, stream>>>(mixb, wfb, bf_, h, fsum, fsq);

    // BN apply
    bn_kernel<<<(int)(XB / 256), 256, 0, stream>>>(h, fsum, fsq, gamma, beta,
                                                   (float*)d_out, (int)XB);
}
